// AttentionStreamBlock_2671469658115
// MI455X (gfx1250) — matrix operand order repaired and checked by reading, not compiled, not run on hardware
//
#include <hip/hip_runtime.h>
#include <hip/hip_bf16.h>

#define D_DIM   1024
#define S_LEN   4096
#define B_SZ    4
#define NTOK    (B_SZ * S_LEN)     /* 16384 tokens */
#define NCHUNK  64                 /* chunks per batch for the scan */
#define CHUNK_L (S_LEN / NCHUNK)   /* 64 steps per chunk */
#define EPS_LN  1e-5f
#define EPS_REC 1e-8f

typedef __attribute__((ext_vector_type(16))) __bf16 v16bf;
typedef __attribute__((ext_vector_type(8)))  __bf16 v8bf;
typedef __attribute__((ext_vector_type(8)))  float  v8f;

// ---------------------------------------------------------------- utilities

__device__ __forceinline__ float sigmoidf_(float x) {
  return 1.0f / (1.0f + expf(-x));
}

// 256-thread (8 wave32) block reduction; smem must hold >= 9 floats.
__device__ __forceinline__ float block_reduce_sum(float v, float* smem) {
#pragma unroll
  for (int m = 16; m > 0; m >>= 1) v += __shfl_xor(v, m, 32);
  const int wid = threadIdx.x >> 5;
  const int lid = threadIdx.x & 31;
  if (lid == 0) smem[wid] = v;
  __syncthreads();
  float r = 0.0f;
  if (threadIdx.x < 8) r = smem[threadIdx.x];
  if (threadIdx.x < 32) {
#pragma unroll
    for (int m = 4; m > 0; m >>= 1) r += __shfl_xor(r, m, 32);
  }
  if (threadIdx.x == 0) smem[8] = r;
  __syncthreads();
  const float out = smem[8];
  __syncthreads();
  return out;
}

__device__ __forceinline__ __bf16 f2bf(float f) { return (__bf16)f; }

// ------------------------------------------------------- 0) weight convert
// fp32 weight -> (hi, lo) bf16 split:  w ~= hi + lo
__global__ __launch_bounds__(256) void k_convert_w(const float* __restrict__ src,
                                                   __bf16* __restrict__ hi,
                                                   __bf16* __restrict__ lo, int n) {
  int i = blockIdx.x * 256 + threadIdx.x;
  if (i < n) {
    float f = src[i];
    __bf16 h = f2bf(f);
    hi[i] = h;
    lo[i] = f2bf(f - (float)h);
  }
}

// --------------------------------------------- 1) per-token elementwise pass
// shifted = mu*x + (1-mu)*x_prev ; d = sigmoid(shifted . decay_w + decay_b)
// ea = sum(exp(shifted)) ; ev = exp(shifted)*shifted
__global__ __launch_bounds__(256) void k_token(const float* __restrict__ x,
                                               const float* __restrict__ xp,
                                               const float* __restrict__ mu_p,
                                               const float* __restrict__ decay_w,
                                               const float* __restrict__ decay_b,
                                               float* __restrict__ evBuf,
                                               float* __restrict__ dBuf,
                                               float* __restrict__ eaBuf) {
  __shared__ float smem[9];
  const int tok = blockIdx.x;
  const float mu = mu_p[0];
  const float om = 1.0f - mu;
  const size_t base = (size_t)tok * D_DIM;

  float sh[4], ex[4];
  float dotAcc = 0.0f, eAcc = 0.0f;
#pragma unroll
  for (int j = 0; j < 4; ++j) {
    const int c = threadIdx.x + j * 256;
    const float v = mu * x[base + c] + om * xp[base + c];
    const float e = expf(v);
    sh[j] = v; ex[j] = e;
    dotAcc += v * decay_w[c];
    eAcc   += e;
  }
  const float dot = block_reduce_sum(dotAcc, smem);
  const float ea  = block_reduce_sum(eAcc, smem);
  if (threadIdx.x == 0) {
    dBuf[tok]  = sigmoidf_(dot + decay_b[0]);
    eaBuf[tok] = ea;
  }
#pragma unroll
  for (int j = 0; j < 4; ++j) {
    const int c = threadIdx.x + j * 256;
    evBuf[base + c] = ex[j] * sh[j];
  }
}

// -------------------------------------------------- 2) scan phase 1 (local)
// Per (batch, chunk): run recurrence from zero carry; emit chunk aggregates.
__global__ __launch_bounds__(256) void k_scan1(const float* __restrict__ evBuf,
                                               const float* __restrict__ dBuf,
                                               const float* __restrict__ eaBuf,
                                               float* __restrict__ chunk_b,
                                               float* __restrict__ chunk_a,
                                               float* __restrict__ chunk_P) {
  __shared__ float ds[CHUNK_L];
  __shared__ float eas[CHUNK_L];
  const int b  = blockIdx.x / NCHUNK;
  const int ch = blockIdx.x % NCHUNK;
  const int t0 = ch * CHUNK_L;
  if (threadIdx.x < CHUNK_L) {
    ds[threadIdx.x]  = dBuf[b * S_LEN + t0 + threadIdx.x];
    eas[threadIdx.x] = eaBuf[b * S_LEN + t0 + threadIdx.x];
  }
  __syncthreads();

  const float* evp = evBuf + ((size_t)b * S_LEN + t0) * D_DIM;
  float bc[4] = {0.f, 0.f, 0.f, 0.f};
  for (int t = 0; t < CHUNK_L; ++t) {
    const float dt = ds[t];
    const size_t r = (size_t)t * D_DIM + threadIdx.x;
#pragma unroll
    for (int j = 0; j < 4; ++j) bc[j] = dt * bc[j] + evp[r + j * 256];
  }
  const size_t cb = (size_t)(b * NCHUNK + ch) * D_DIM + threadIdx.x;
#pragma unroll
  for (int j = 0; j < 4; ++j) chunk_b[cb + j * 256] = bc[j];

  if (threadIdx.x == 0) {
    float a = 0.f, P = 1.f;
    for (int t = 0; t < CHUNK_L; ++t) { a = ds[t] * a + eas[t]; P *= ds[t]; }
    chunk_a[b * NCHUNK + ch] = a;
    chunk_P[b * NCHUNK + ch] = P;
  }
}

// --------------------------------------- 3) scan phase 2 (cross-chunk scan)
__global__ __launch_bounds__(256) void k_scan2(const float* __restrict__ chunk_b,
                                               const float* __restrict__ chunk_a,
                                               const float* __restrict__ chunk_P,
                                               float* __restrict__ carry_b,
                                               float* __restrict__ carry_a) {
  const int b = blockIdx.x;
  float cb[4] = {0.f, 0.f, 0.f, 0.f};
  float ca = 0.f;
  for (int ch = 0; ch < NCHUNK; ++ch) {
    const int cidx = b * NCHUNK + ch;
    const size_t off = (size_t)cidx * D_DIM + threadIdx.x;
#pragma unroll
    for (int j = 0; j < 4; ++j) carry_b[off + j * 256] = cb[j];
    if (threadIdx.x == 0) carry_a[cidx] = ca;
    const float P = chunk_P[cidx];
    const float A = chunk_a[cidx];
#pragma unroll
    for (int j = 0; j < 4; ++j) cb[j] = P * cb[j] + chunk_b[off + j * 256];
    ca = P * ca + A;
  }
}

// -------------------------------- 4) scan phase 3 (apply carries, in place)
// Overwrites evBuf with scan output b_t / (a_t + eps)
__global__ __launch_bounds__(256) void k_scan3(float* __restrict__ evBuf,
                                               const float* __restrict__ dBuf,
                                               const float* __restrict__ eaBuf,
                                               const float* __restrict__ carry_b,
                                               const float* __restrict__ carry_a) {
  __shared__ float ds[CHUNK_L];
  __shared__ float eas[CHUNK_L];
  const int b  = blockIdx.x / NCHUNK;
  const int ch = blockIdx.x % NCHUNK;
  const int t0 = ch * CHUNK_L;
  if (threadIdx.x < CHUNK_L) {
    ds[threadIdx.x]  = dBuf[b * S_LEN + t0 + threadIdx.x];
    eas[threadIdx.x] = eaBuf[b * S_LEN + t0 + threadIdx.x];
  }
  __syncthreads();

  const int cidx = b * NCHUNK + ch;
  float a = carry_a[cidx];
  float bc[4];
  const size_t coff = (size_t)cidx * D_DIM + threadIdx.x;
#pragma unroll
  for (int j = 0; j < 4; ++j) bc[j] = carry_b[coff + j * 256];

  float* evp = evBuf + ((size_t)b * S_LEN + t0) * D_DIM;
  for (int t = 0; t < CHUNK_L; ++t) {
    const float dt = ds[t];
    a = dt * a + eas[t];
    const float inv = 1.0f / (a + EPS_REC);
    const size_t r = (size_t)t * D_DIM + threadIdx.x;
#pragma unroll
    for (int j = 0; j < 4; ++j) {
      bc[j] = dt * bc[j] + evp[r + j * 256];
      evp[r + j * 256] = bc[j] * inv;
    }
  }
}

// ----------------------------------------------- 5) LN1 -> merged hi/lo bf16
__global__ __launch_bounds__(256) void k_ln1(const float* __restrict__ scanBuf,
                                             const float* __restrict__ g,
                                             const float* __restrict__ bta,
                                             __bf16* __restrict__ mHi,
                                             __bf16* __restrict__ mLo) {
  __shared__ float smem[9];
  const int tok = blockIdx.x;
  const size_t base = (size_t)tok * D_DIM;
  float v[4];
  float s = 0.f, sq = 0.f;
#pragma unroll
  for (int j = 0; j < 4; ++j) {
    const int c = threadIdx.x + j * 256;
    v[j] = scanBuf[base + c];
    s += v[j]; sq += v[j] * v[j];
  }
  const float sum  = block_reduce_sum(s, smem);
  const float sumq = block_reduce_sum(sq, smem);
  const float mean = sum * (1.0f / D_DIM);
  const float var  = sumq * (1.0f / D_DIM) - mean * mean;
  const float rstd = rsqrtf(var + EPS_LN);
#pragma unroll
  for (int j = 0; j < 4; ++j) {
    const int c = threadIdx.x + j * 256;
    const float m = (v[j] - mean) * rstd * g[c] + bta[c];
    const __bf16 h = f2bf(m);
    mHi[base + c] = h;
    mLo[base + c] = f2bf(m - (float)h);
  }
}

// ------------------------------------------------------ 6) dual WMMA GEMM
// gate = sigmoid(merged @ Wr^T + Wr_b), value = merged @ Wv^T + Wv_b,
// writes gate*value into `coupled`.
// Register blocking: each wave computes TWO 16x16 output tiles (rows m0 and
// m0+16) for one 16-column tile, so each weight fragment feeds 2 WMMAs.
// Per k-step: 16 b128 loads -> 12 v_wmma_f32_16x16x32_bf16.
// Split precision: hiA*hiB + hiA*loB + loA*hiB (~fp32 fidelity).

__device__ __forceinline__ v16bf load_frag16(const __bf16* p0, const __bf16* p1) {
  v8bf a = *(const v8bf*)p0;
  v8bf b = *(const v8bf*)p1;
  return __builtin_shufflevector(a, b, 0, 1, 2, 3, 4, 5, 6, 7,
                                 8, 9, 10, 11, 12, 13, 14, 15);
}

#define WMMA_BF16(A, Bf, C) \
  __builtin_amdgcn_wmma_f32_16x16x32_bf16(false, (A), false, (Bf), (short)0, (C), false, false)

__global__ __launch_bounds__(256) void k_gemm(const __bf16* __restrict__ mHi,
                                              const __bf16* __restrict__ mLo,
                                              const __bf16* __restrict__ WrHi,
                                              const __bf16* __restrict__ WrLo,
                                              const __bf16* __restrict__ WvHi,
                                              const __bf16* __restrict__ WvLo,
                                              const float* __restrict__ Wr_b,
                                              const float* __restrict__ Wv_b,
                                              float* __restrict__ coupled) {
  const int wave = threadIdx.x >> 5;
  const int lane = threadIdx.x & 31;
  const int m0 = blockIdx.y * 32;                 // two row tiles: m0, m0+16
  const int n0 = (blockIdx.x * 8 + wave) * 16;    // output-channel tile base
  const int hg = lane >> 4;                       // half-group (0 or 1)
  const int lm = lane & 15;

  // A fragments: row = tile+lm, halves 0-7 at k+hg*8, halves 8-15 at k+16+hg*8
  const __bf16* aHi0 = mHi + (size_t)(m0 + lm) * D_DIM + hg * 8;
  const __bf16* aLo0 = mLo + (size_t)(m0 + lm) * D_DIM + hg * 8;
  const __bf16* aHi1 = aHi0 + (size_t)16 * D_DIM;
  const __bf16* aLo1 = aLo0 + (size_t)16 * D_DIM;
  // B fragments: weight row n0+lm, halves 0-7 at k+hg*8, halves 8-15 at k+16+hg*8
  const __bf16* brHi = WrHi + (size_t)(n0 + lm) * D_DIM + hg * 8;
  const __bf16* brLo = WrLo + (size_t)(n0 + lm) * D_DIM + hg * 8;
  const __bf16* bvHi = WvHi + (size_t)(n0 + lm) * D_DIM + hg * 8;
  const __bf16* bvLo = WvLo + (size_t)(n0 + lm) * D_DIM + hg * 8;

  v8f accG0 = {0.f, 0.f, 0.f, 0.f, 0.f, 0.f, 0.f, 0.f};
  v8f accV0 = {0.f, 0.f, 0.f, 0.f, 0.f, 0.f, 0.f, 0.f};
  v8f accG1 = {0.f, 0.f, 0.f, 0.f, 0.f, 0.f, 0.f, 0.f};
  v8f accV1 = {0.f, 0.f, 0.f, 0.f, 0.f, 0.f, 0.f, 0.f};

  for (int k = 0; k < D_DIM; k += 32) {
    const v16bf a0h = load_frag16(aHi0 + k, aHi0 + k + 16);
    const v16bf a0l = load_frag16(aLo0 + k, aLo0 + k + 16);
    const v16bf a1h = load_frag16(aHi1 + k, aHi1 + k + 16);
    const v16bf a1l = load_frag16(aLo1 + k, aLo1 + k + 16);
    const v16bf bGH = load_frag16(brHi + k, brHi + k + 16);
    const v16bf bGL = load_frag16(brLo + k, brLo + k + 16);
    const v16bf bVH = load_frag16(bvHi + k, bvHi + k + 16);
    const v16bf bVL = load_frag16(bvLo + k, bvLo + k + 16);

    accG0 = WMMA_BF16(a0h, bGH, accG0);
    accG0 = WMMA_BF16(a0h, bGL, accG0);
    accG0 = WMMA_BF16(a0l, bGH, accG0);
    accG1 = WMMA_BF16(a1h, bGH, accG1);
    accG1 = WMMA_BF16(a1h, bGL, accG1);
    accG1 = WMMA_BF16(a1l, bGH, accG1);

    accV0 = WMMA_BF16(a0h, bVH, accV0);
    accV0 = WMMA_BF16(a0h, bVL, accV0);
    accV0 = WMMA_BF16(a0l, bVH, accV0);
    accV1 = WMMA_BF16(a1h, bVH, accV1);
    accV1 = WMMA_BF16(a1h, bVL, accV1);
    accV1 = WMMA_BF16(a1l, bVH, accV1);
  }

  // C/D layout: VGPR j -> row tile + hg*8 + j, col n0 + lm
  const int col = n0 + lm;
  const float rb = Wr_b[col];
  const float vb = Wv_b[col];
#pragma unroll
  for (int j = 0; j < 8; ++j) {
    const int r0 = m0 + hg * 8 + j;
    const float g0 = sigmoidf_(accG0[j] + rb);
    coupled[(size_t)r0 * D_DIM + col] = g0 * (accV0[j] + vb);
    const int r1 = m0 + 16 + hg * 8 + j;
    const float g1 = sigmoidf_(accG1[j] + rb);
    coupled[(size_t)r1 * D_DIM + col] = g1 * (accV1[j] + vb);
  }
}

// --------------------------------------------- 7) LN2 + residual (final out)
__global__ __launch_bounds__(256) void k_ln2_final(const float* __restrict__ coupled,
                                                   const float* __restrict__ x,
                                                   const float* __restrict__ xp,
                                                   const float* __restrict__ mu_p,
                                                   const float* __restrict__ g,
                                                   const float* __restrict__ bta,
                                                   float* __restrict__ out) {
  __shared__ float smem[9];
  const int tok = blockIdx.x;
  const size_t base = (size_t)tok * D_DIM;
  const float mu = mu_p[0];
  const float om = 1.0f - mu;
  float v[4];
  float s = 0.f, sq = 0.f;
#pragma unroll
  for (int j = 0; j < 4; ++j) {
    const int c = threadIdx.x + j * 256;
    v[j] = coupled[base + c];
    s += v[j]; sq += v[j] * v[j];
  }
  const float sum  = block_reduce_sum(s, smem);
  const float sumq = block_reduce_sum(sq, smem);
  const float mean = sum * (1.0f / D_DIM);
  const float var  = sumq * (1.0f / D_DIM) - mean * mean;
  const float rstd = rsqrtf(var + EPS_LN);
#pragma unroll
  for (int j = 0; j < 4; ++j) {
    const int c = threadIdx.x + j * 256;
    const float shifted = mu * x[base + c] + om * xp[base + c];
    out[base + c] = shifted + (v[j] - mean) * rstd * g[c] + bta[c];
  }
}

// --------------------------------------------------------------- launcher

extern "C" void kernel_launch(void* const* d_in, const int* in_sizes, int n_in,
                              void* d_out, int out_size, void* d_ws, size_t ws_size,
                              hipStream_t stream) {
  const float* x       = (const float*)d_in[0];
  const float* x_prev  = (const float*)d_in[1];
  const float* mu      = (const float*)d_in[2];
  const float* decay_w = (const float*)d_in[3];
  const float* decay_b = (const float*)d_in[4];
  const float* ln1_g   = (const float*)d_in[5];
  const float* ln1_b   = (const float*)d_in[6];
  const float* Wr_w    = (const float*)d_in[7];
  const float* Wr_b    = (const float*)d_in[8];
  const float* Wv_w    = (const float*)d_in[9];
  const float* Wv_b    = (const float*)d_in[10];
  const float* ln2_g   = (const float*)d_in[11];
  const float* ln2_b   = (const float*)d_in[12];
  float* out = (float*)d_out;

  // bump allocator over workspace (256B aligned)
  char* wp = (char*)d_ws;
  auto alloc = [&](size_t bytes) -> char* {
    char* p = wp;
    wp += (bytes + 255) & ~(size_t)255;
    return p;
  };
  const size_t tokElems = (size_t)NTOK * D_DIM;

  float*  evBuf   = (float*)alloc(tokElems * sizeof(float));  // ev -> scan -> coupled
  __bf16* mHi     = (__bf16*)alloc(tokElems * sizeof(__bf16));
  __bf16* mLo     = (__bf16*)alloc(tokElems * sizeof(__bf16));
  __bf16* WrHi    = (__bf16*)alloc((size_t)D_DIM * D_DIM * sizeof(__bf16));
  __bf16* WrLo    = (__bf16*)alloc((size_t)D_DIM * D_DIM * sizeof(__bf16));
  __bf16* WvHi    = (__bf16*)alloc((size_t)D_DIM * D_DIM * sizeof(__bf16));
  __bf16* WvLo    = (__bf16*)alloc((size_t)D_DIM * D_DIM * sizeof(__bf16));
  float*  dBuf    = (float*)alloc((size_t)NTOK * sizeof(float));
  float*  eaBuf   = (float*)alloc((size_t)NTOK * sizeof(float));
  float*  chunk_b = (float*)alloc((size_t)B_SZ * NCHUNK * D_DIM * sizeof(float));
  float*  carry_b = (float*)alloc((size_t)B_SZ * NCHUNK * D_DIM * sizeof(float));
  float*  chunk_a = (float*)alloc((size_t)B_SZ * NCHUNK * sizeof(float));
  float*  chunk_P = (float*)alloc((size_t)B_SZ * NCHUNK * sizeof(float));
  float*  carry_a = (float*)alloc((size_t)B_SZ * NCHUNK * sizeof(float));
  (void)ws_size; (void)in_sizes; (void)n_in; (void)out_size;

  const int wN = D_DIM * D_DIM;
  k_convert_w<<<(wN + 255) / 256, 256, 0, stream>>>(Wr_w, WrHi, WrLo, wN);
  k_convert_w<<<(wN + 255) / 256, 256, 0, stream>>>(Wv_w, WvHi, WvLo, wN);

  k_token<<<NTOK, 256, 0, stream>>>(x, x_prev, mu, decay_w, decay_b,
                                    evBuf, dBuf, eaBuf);

  k_scan1<<<B_SZ * NCHUNK, 256, 0, stream>>>(evBuf, dBuf, eaBuf,
                                             chunk_b, chunk_a, chunk_P);
  k_scan2<<<B_SZ, 256, 0, stream>>>(chunk_b, chunk_a, chunk_P, carry_b, carry_a);
  k_scan3<<<B_SZ * NCHUNK, 256, 0, stream>>>(evBuf, dBuf, eaBuf, carry_b, carry_a);

  k_ln1<<<NTOK, 256, 0, stream>>>(evBuf, ln1_g, ln1_b, mHi, mLo);

  {
    dim3 grid(D_DIM / 128, NTOK / 32);  // (8, 512)
    k_gemm<<<grid, 256, 0, stream>>>(mHi, mLo, WrHi, WrLo, WvHi, WvLo,
                                     Wr_b, Wv_b, evBuf);
  }

  k_ln2_final<<<NTOK, 256, 0, stream>>>(evBuf, x, x_prev, mu, ln2_g, ln2_b, out);
}
